// Atom_Atom_embedding_MP_15942918603399
// MI455X (gfx1250) — matrix operand order repaired and checked by reading, compiled, not run on hardware
//
#include <hip/hip_runtime.h>

#define D_FEAT  64
#define K_NBR   16
#define H_DIM   129
#define HP      160   // K-dim padded to 5 chunks of 32
#define NP1     144   // W1 output cols padded to 9 tiles of 16
#define NT1     9
#define KC      5
#define NT2     4
#define NLAYERS 3
#define WAVES   4
#define SLOPE   0.2f
#define EPS_GN  1e-5f

typedef __attribute__((ext_vector_type(16))) _Float16 v16h;
typedef __attribute__((ext_vector_type(8)))  float    v8f;
typedef __attribute__((ext_vector_type(4)))  unsigned v4u;

union AFrag { v16h v; unsigned u[8]; };
union BFrag { v16h v; v4u q[2]; unsigned u[8]; };
union CAcc  { v8f  v; float f[8]; };
union PackH { _Float16 h[2]; unsigned u; };

// ---------------- weight pre-pack (f32 -> padded f16) ----------------

__global__ void pack_w1(const float* __restrict__ w1, _Float16* __restrict__ w1h) {
    int i = blockIdx.x * blockDim.x + threadIdx.x;
    const int total = NLAYERS * HP * NP1;
    if (i >= total) return;
    int l = i / (HP * NP1);
    int r = i % (HP * NP1);
    int h = r / NP1;          // K (input) index
    int m = r % NP1;          // N (output) index
    float v = (h < H_DIM && m < H_DIM) ? w1[(l * H_DIM + h) * H_DIM + m] : 0.0f;
    w1h[i] = (_Float16)v;
}

__global__ void pack_w2(const float* __restrict__ w2, _Float16* __restrict__ w2h) {
    int i = blockIdx.x * blockDim.x + threadIdx.x;
    const int total = NLAYERS * HP * D_FEAT;
    if (i >= total) return;
    int l = i / (HP * D_FEAT);
    int r = i % (HP * D_FEAT);
    int h = r / D_FEAT;
    int d = r % D_FEAT;
    float v = (h < H_DIM) ? w2[(l * H_DIM + h) * D_FEAT + d] : 0.0f;
    w2h[i] = (_Float16)v;
}

// ---------------- one message-passing layer, one wave per node ----------------

__global__ __launch_bounds__(WAVES * 32)
void layer_kernel(const float* __restrict__ xin,
                  const float* __restrict__ dists,
                  const long long* __restrict__ idx,
                  const _Float16* __restrict__ w1h,   // [HP][NP1] f16, this layer
                  const float* __restrict__ b1,       // [H_DIM]
                  const _Float16* __restrict__ w2h,   // [HP][D_FEAT] f16, this layer
                  const float* __restrict__ b2,       // [D_FEAT]
                  const float* __restrict__ gamma,
                  const float* __restrict__ beta,
                  float* __restrict__ xout,
                  int nNodes)
{
    __shared__ unsigned  sFeat[WAVES][K_NBR][HP / 2];   // feats as packed f16 pairs
    __shared__ _Float16  sH[WAVES][K_NBR][HP];          // hidden activations f16
    __shared__ float     sMsg[WAVES][D_FEAT];

    const int wave = threadIdx.x >> 5;
    const int lane = threadIdx.x & 31;
    const int hi   = lane >> 4;      // which half-wave
    const int ml   = lane & 15;      // M row (lanes 0-15 / 16-31 mirror)

    long long node = (long long)blockIdx.x * WAVES + wave;
    if (node >= nNodes) node = nNodes - 1;   // wave-uniform clamp: EXEC stays all-1s

    // ---- stage feats = [self(64) | nbr(64) | dist | 0-pad] as f16 in LDS ----
    const float s0 = xin[node * D_FEAT + 2 * lane];
    const float s1 = xin[node * D_FEAT + 2 * lane + 1];
    PackH ps; ps.h[0] = (_Float16)s0; ps.h[1] = (_Float16)s1;
    #pragma unroll
    for (int k = 0; k < K_NBR; ++k)
        sFeat[wave][k][lane] = ps.u;                       // cols [0,64)

    #pragma unroll
    for (int k = 0; k < K_NBR; ++k) {
        long long j = idx[node * K_NBR + k];
        float n0 = xin[j * D_FEAT + 2 * lane];
        float n1 = xin[j * D_FEAT + 2 * lane + 1];
        PackH pn; pn.h[0] = (_Float16)n0; pn.h[1] = (_Float16)n1;
        sFeat[wave][k][32 + lane] = pn.u;                  // cols [64,128)
        float dk = dists[node * K_NBR + k];
        PackH pd; pd.h[0] = (_Float16)dk; pd.h[1] = (_Float16)0.0f;
        if (lane < 16)
            sFeat[wave][k][64 + lane] = (lane == 0) ? pd.u : 0u;  // col 128 = dist, 129..159 = 0
    }

    // ---- GEMM1: h = leaky(feats[16x129] @ W1[129x129] + b1), via 16x16x32 f16 WMMA ----
    const unsigned* fU = &sFeat[wave][0][0];   // [16][HP/2]
    CAcc acc[NT1];
    #pragma unroll
    for (int t = 0; t < NT1; ++t)
        #pragma unroll
        for (int r = 0; r < 8; ++r) acc[t].f[r] = 0.0f;

    #pragma unroll
    for (int kc = 0; kc < KC; ++kc) {
        // A-fragment: lane (hi,ml): VGPR v holds K pair at base {0|16} + hi*8 + (v&3)*2
        AFrag a;
        #pragma unroll
        for (int vv = 0; vv < 8; ++vv) {
            int kp = kc * 16 + ((vv < 4) ? 0 : 8) + hi * 4 + (vv & 3);  // u32 index in row
            a.u[vv] = fU[ml * (HP / 2) + kp];
        }
        #pragma unroll
        for (int t = 0; t < NT1; ++t) {
            BFrag b;
            #pragma unroll
            for (int s = 0; s < 16; ++s) {
                int kk = kc * 32 + ((s < 8) ? 0 : 8) + hi * 8 + s;
                b.v[s] = w1h[(size_t)kk * NP1 + t * 16 + ml];
            }
            acc[t].v = __builtin_amdgcn_wmma_f32_16x16x32_f16(
                false, a.v, false, b.v, (short)0, acc[t].v, false, false);
        }
    }

    // epilogue: bias + leaky-relu, store f16 to LDS; zero pad cols [144,160)
    _Float16* hB = &sH[wave][0][0];
    unsigned* hU = (unsigned*)hB;
    #pragma unroll
    for (int t = 0; t < 4; ++t) {
        int i = t * 32 + lane;                 // 128 pad u32's: rows 0..15, u32 cols 72..79
        hU[(i >> 3) * (HP / 2) + 72 + (i & 7)] = 0u;
    }
    #pragma unroll
    for (int t = 0; t < NT1; ++t) {
        int col = t * 16 + ml;
        float bias = (col < H_DIM) ? b1[col] : 0.0f;
        #pragma unroll
        for (int r = 0; r < 8; ++r) {
            float x = acc[t].f[r] + bias;
            x = (x >= 0.0f) ? x : SLOPE * x;
            hB[(r + hi * 8) * HP + col] = (_Float16)x;
        }
    }

    // ---- GEMM2: msgK = h[16x129] @ W2[129x64]  (then sum over the 16 rows) ----
    CAcc acc2[NT2];
    #pragma unroll
    for (int t = 0; t < NT2; ++t)
        #pragma unroll
        for (int r = 0; r < 8; ++r) acc2[t].f[r] = 0.0f;

    const unsigned* hUc = (const unsigned*)hB;
    #pragma unroll
    for (int kc = 0; kc < KC; ++kc) {
        AFrag a;
        #pragma unroll
        for (int vv = 0; vv < 8; ++vv) {
            int kp = kc * 16 + ((vv < 4) ? 0 : 8) + hi * 4 + (vv & 3);
            a.u[vv] = hUc[ml * (HP / 2) + kp];
        }
        #pragma unroll
        for (int t = 0; t < NT2; ++t) {
            BFrag b;
            #pragma unroll
            for (int s = 0; s < 16; ++s) {
                int kk = kc * 32 + ((s < 8) ? 0 : 8) + hi * 8 + s;
                b.v[s] = w2h[(size_t)kk * D_FEAT + t * 16 + ml];
            }
            acc2[t].v = __builtin_amdgcn_wmma_f32_16x16x32_f16(
                false, a.v, false, b.v, (short)0, acc2[t].v, false, false);
        }
    }

    // ---- sum over K: rows within lane, then lane<->lane+16 ----
    #pragma unroll
    for (int t = 0; t < NT2; ++t) {
        float p = 0.0f;
        #pragma unroll
        for (int r = 0; r < 8; ++r) p += acc2[t].f[r];
        p += __shfl_xor(p, 16, 32);
        if (lane < 16) sMsg[wave][t * 16 + lane] = p;
    }

    // ---- group norm over D=64 (2 values per lane) + leaky + residual ----
    float x0 = sMsg[wave][lane]      + 16.0f * b2[lane];        // + K * b2 (bias summed K times)
    float x1 = sMsg[wave][32 + lane] + 16.0f * b2[32 + lane];
    float s = x0 + x1;
    float q = x0 * x0 + x1 * x1;
    #pragma unroll
    for (int o = 16; o >= 1; o >>= 1) {
        s += __shfl_xor(s, o, 32);
        q += __shfl_xor(q, o, 32);
    }
    float mean = s * (1.0f / 64.0f);
    float var  = q * (1.0f / 64.0f) - mean * mean;
    float rs   = rsqrtf(var + EPS_GN);
    float y0 = (x0 - mean) * rs * gamma[lane]      + beta[lane];
    float y1 = (x1 - mean) * rs * gamma[32 + lane] + beta[32 + lane];
    y0 = (y0 >= 0.0f) ? y0 : SLOPE * y0;
    y1 = (y1 >= 0.0f) ? y1 : SLOPE * y1;
    xout[node * D_FEAT + lane]      = xin[node * D_FEAT + lane]      + y0;
    xout[node * D_FEAT + 32 + lane] = xin[node * D_FEAT + 32 + lane] + y1;
}

// ---------------- host launch ----------------

extern "C" void kernel_launch(void* const* d_in, const int* in_sizes, int n_in,
                              void* d_out, int out_size, void* d_ws, size_t ws_size,
                              hipStream_t stream) {
    const float*     y     = (const float*)d_in[0];
    const float*     dist  = (const float*)d_in[1];
    const float*     w1    = (const float*)d_in[2];
    const float*     b1    = (const float*)d_in[3];
    const float*     w2    = (const float*)d_in[4];
    const float*     b2    = (const float*)d_in[5];
    const float*     gam   = (const float*)d_in[6];
    const float*     bet   = (const float*)d_in[7];
    const long long* idx   = (const long long*)d_in[8];   // jnp.int64 in reference

    const int nNodes = in_sizes[0] / D_FEAT;

    char* ws = (char*)d_ws;
    _Float16* w1h = (_Float16*)ws;                                   // 3*160*144*2 B
    _Float16* w2h = (_Float16*)(ws + (size_t)NLAYERS * HP * NP1 * 2); // 3*160*64*2 B
    size_t wbytes = (size_t)NLAYERS * HP * NP1 * 2 + (size_t)NLAYERS * HP * D_FEAT * 2;
    size_t bufOff = (wbytes + 255) & ~(size_t)255;
    float* buf0 = (float*)(ws + bufOff);
    float* buf1 = (float*)(ws + bufOff + (size_t)nNodes * D_FEAT * sizeof(float));

    {
        int tot1 = NLAYERS * HP * NP1;
        pack_w1<<<(tot1 + 255) / 256, 256, 0, stream>>>(w1, w1h);
        int tot2 = NLAYERS * HP * D_FEAT;
        pack_w2<<<(tot2 + 255) / 256, 256, 0, stream>>>(w2, w2h);
    }

    const int blocks = (nNodes + WAVES - 1) / WAVES;
    const float* cin = y;
    float* couts[NLAYERS] = { buf0, buf1, (float*)d_out };
    for (int l = 0; l < NLAYERS; ++l) {
        layer_kernel<<<blocks, WAVES * 32, 0, stream>>>(
            cin, dist, idx,
            w1h + (size_t)l * HP * NP1,    b1 + (size_t)l * H_DIM,
            w2h + (size_t)l * HP * D_FEAT, b2 + (size_t)l * D_FEAT,
            gam + (size_t)l * D_FEAT,      bet + (size_t)l * D_FEAT,
            couts[l], nNodes);
        cin = couts[l];
    }
}
